// CoreBase_82686710382599
// MI455X (gfx1250) — matrix operand order repaired and checked by reading, not compiled, not run on hardware
//
#include <hip/hip_runtime.h>

// ---------------- constants ----------------
#define DD   512
#define FFD  1024
#define BSZ  512
#define LP   144     // padded molecule sequence length (139 valid)
#define LMV  139     // valid molecule sequence length
#define LPK  160     // padded K-dim for attention P matrix (cols >=144 zero)
#define LR   8       // ring sequence length (max ring size)

typedef __attribute__((ext_vector_type(16))) _Float16 v16h;
typedef __attribute__((ext_vector_type(8)))  float    v8f;

union FragH {
    v16h v;
    uint4 q[2];
    _Float16 e[16];
};

__device__ inline float h2f(_Float16 h) { return (float)h; }

// CDNA5 async global->LDS copy (ASYNCcnt-tracked), see CDNA5 ISA
__device__ inline void async_copy_b128(unsigned lds_byte_off, const void* gaddr) {
    asm volatile("global_load_async_to_lds_b128 %0, %1, off"
                 :: "v"(lds_byte_off), "v"(gaddr) : "memory");
}
__device__ inline void wait_async0() {
    asm volatile("s_wait_asynccnt 0x0" ::: "memory");
}

// ---------------- tiny utility kernels ----------------
__global__ void k_f32_to_f16(const float* __restrict__ s, _Float16* __restrict__ d, size_t n) {
    size_t i = (size_t)blockIdx.x * 256 + threadIdx.x;
    if (i < n) d[i] = (_Float16)s[i];
}

__global__ void k_scan_excl(const int* __restrict__ in, int* __restrict__ out, int n) {
    if (blockIdx.x == 0 && threadIdx.x == 0) {
        int s = 0;
        for (int i = 0; i < n; ++i) { out[i] = s; s += in[i]; }
    }
}

// ---------------- stage builders ----------------
__global__ void k_build_ring_tokens(const float* __restrict__ x,
                                    const int* __restrict__ rni,
                                    const int* __restrict__ rnn,
                                    const int* __restrict__ rstart,
                                    float* __restrict__ h0,
                                    _Float16* __restrict__ hb) {
    int t = blockIdx.x;           // r*8 + j
    int r = t >> 3, j = t & 7;
    int d = threadIdx.x;          // 512 threads
    float v = 0.f;
    if (j < rnn[r]) {
        int src = rni[rstart[r] + j];
        v = x[(size_t)src * DD + d];
    }
    size_t o = (size_t)t * DD + d;
    h0[o] = v;
    hb[o] = (_Float16)v;
}

__global__ void k_build_seq(const float* __restrict__ x,
                            const int* __restrict__ ptr,
                            const int* __restrict__ mrn,
                            const int* __restrict__ rstartB,
                            const float* __restrict__ CLS,
                            const float* __restrict__ RING,
                            const float* __restrict__ END,
                            const float* __restrict__ rings_vec,
                            float* __restrict__ seqf,
                            _Float16* __restrict__ seqh) {
    int b = blockIdx.x / LP, l = blockIdx.x % LP;
    int d = threadIdx.x;          // 512 threads
    float v = 0.f;
    if (l == 0) v = CLS[d];
    else if (l <= 128) {
        int i = l - 1;
        int nn = ptr[b + 1] - ptr[b];
        if (i < nn) v = x[((size_t)ptr[b] + i) * DD + d];
    } else if (l == 129) v = RING[d];
    else if (l <= 137) {
        int j = l - 130;
        if (j < mrn[b]) v = rings_vec[((size_t)rstartB[b] + j) * DD + d];
    } else if (l == 138) v = END[d];
    size_t o = ((size_t)b * LP + l) * DD + d;
    seqf[o] = v;
    seqh[o] = (_Float16)v;
}

__global__ void k_build_kpm(const int* __restrict__ ptr,
                            const int* __restrict__ mrn,
                            unsigned char* __restrict__ kpm) {
    int b = blockIdx.x;
    int l = threadIdx.x;          // 256 threads, guard
    if (l < LP) {
        int nn = ptr[b + 1] - ptr[b];
        bool m = false;
        if (l >= 1 && l <= 128)       m = (l - 1) >= nn;
        else if (l >= 130 && l <= 137) m = (l - 130) >= mrn[b];
        else if (l >= LMV)             m = true;
        kpm[b * LP + l] = m ? 1 : 0;
    }
}

// ---------------- WMMA GEMM: C[T x N] = A[T x K] * W[N x K]^T + bias -------------
// Block: 256 thr (8 waves). Macro tile: M=128 (wave m-slices) x N=64.
// B tile (64x32 f16 = 4KB) staged in LDS via async copies, double-buffered.
// Each wave: one A fragment per k-step reused over 4 N-tiles (4 accumulators).
template <bool RELU, bool OUT_F16>
__global__ __launch_bounds__(256) void k_gemm_wmma(
        const _Float16* __restrict__ A, int lda,
        const _Float16* __restrict__ W,
        const float* __restrict__ bias,
        float* __restrict__ Cf,
        _Float16* __restrict__ Ch, int ldc,
        int T, int K) {
    __shared__ __align__(16) _Float16 Bs[2][64 * 32];

    int tid = threadIdx.x;
    int lane = tid & 31, wave = tid >> 5;
    int half = lane >> 4, r = lane & 15;

    int tm = blockIdx.x * 128 + wave * 16;
    int tnb = blockIdx.y * 64;

    int arow = tm + r; if (arow >= T) arow = T - 1;
    const _Float16* aptr = A + (size_t)arow * lda;

    // B staging: each thread owns one 16-byte segment of the 64x32 tile
    int brow = tid >> 2, bseg = tid & 3;                  // row 0..63, seg 0..3
    const _Float16* bg = W + (size_t)(tnb + brow) * K + bseg * 8;
    unsigned bs[2];
    bs[0] = (unsigned)(unsigned long long)&Bs[0][brow * 32 + bseg * 8];
    bs[1] = (unsigned)(unsigned long long)&Bs[1][brow * 32 + bseg * 8];

    v8f acc0 = {}, acc1 = {}, acc2 = {}, acc3 = {};

    int nk = K >> 5;
    async_copy_b128(bs[0], bg);                            // preload k-step 0
    for (int ks = 0; ks < nk; ++ks) {
        int k = ks << 5;
        wait_async0();
        __syncthreads();                                   // buf[ks&1] ready for all waves
        if (ks + 1 < nk)
            async_copy_b128(bs[(ks + 1) & 1], bg + (k + 32));

        FragH fa;
        fa.q[0] = *(const uint4*)(aptr + k + half * 16);
        fa.q[1] = *(const uint4*)(aptr + k + half * 16 + 8);
        if (k + 32 < K) __builtin_prefetch(aptr + k + 32, 0, 0);

        const _Float16* bb = &Bs[ks & 1][0];
        FragH fb;
        fb.q[0] = *(const uint4*)(bb + (0 * 16 + r) * 32 + half * 16);
        fb.q[1] = *(const uint4*)(bb + (0 * 16 + r) * 32 + half * 16 + 8);
        acc0 = __builtin_amdgcn_wmma_f32_16x16x32_f16(false, fa.v, false, fb.v, (short)0, acc0, false, false);
        fb.q[0] = *(const uint4*)(bb + (1 * 16 + r) * 32 + half * 16);
        fb.q[1] = *(const uint4*)(bb + (1 * 16 + r) * 32 + half * 16 + 8);
        acc1 = __builtin_amdgcn_wmma_f32_16x16x32_f16(false, fa.v, false, fb.v, (short)0, acc1, false, false);
        fb.q[0] = *(const uint4*)(bb + (2 * 16 + r) * 32 + half * 16);
        fb.q[1] = *(const uint4*)(bb + (2 * 16 + r) * 32 + half * 16 + 8);
        acc2 = __builtin_amdgcn_wmma_f32_16x16x32_f16(false, fa.v, false, fb.v, (short)0, acc2, false, false);
        fb.q[0] = *(const uint4*)(bb + (3 * 16 + r) * 32 + half * 16);
        fb.q[1] = *(const uint4*)(bb + (3 * 16 + r) * 32 + half * 16 + 8);
        acc3 = __builtin_amdgcn_wmma_f32_16x16x32_f16(false, fa.v, false, fb.v, (short)0, acc3, false, false);

        __syncthreads();                                   // all reads of buf[ks&1] done
    }

    int rowbase = tm + half * 8;
#pragma unroll
    for (int j = 0; j < 4; ++j) {
        v8f acc = (j == 0) ? acc0 : (j == 1) ? acc1 : (j == 2) ? acc2 : acc3;
        int col = tnb + j * 16 + r;
        float bv = bias ? bias[col] : 0.f;
#pragma unroll
        for (int v = 0; v < 8; ++v) {
            int row = rowbase + v;
            if (row < T) {
                float val = acc[v] + bv;
                if (RELU) val = fmaxf(val, 0.f);
                if (OUT_F16) Ch[(size_t)row * ldc + col] = (_Float16)val;
                else         Cf[(size_t)row * ldc + col] = val;
            }
        }
    }
}

// ---------------- molecule attention ----------------
// S = Q K^T * scale + mask   (grid = B*4*9, block = 288 = 9 waves, wave = k-tile)
__global__ void k_attn_scores(const _Float16* __restrict__ qkv,
                              const unsigned char* __restrict__ kpm,
                              float* __restrict__ S) {
    int lane = threadIdx.x & 31;
    int ktile = threadIdx.x >> 5;     // 0..8
    int qtile = blockIdx.x % 9;
    int bh = blockIdx.x / 9;
    int h = bh & 3, b = bh >> 2;
    int half = lane >> 4, r = lane & 15;

    const _Float16* Qb = qkv + (size_t)(b * LP) * 1536 + h * 128;
    const _Float16* Kb = qkv + (size_t)(b * LP) * 1536 + 512 + h * 128;
    const _Float16* aptr = Qb + (size_t)(qtile * 16 + r) * 1536;
    const _Float16* bptr = Kb + (size_t)(ktile * 16 + r) * 1536 + half * 16;

    v8f acc = {};
#pragma unroll
    for (int k = 0; k < 128; k += 32) {
        FragH fa, fb;
        fa.q[0] = *(const uint4*)(aptr + k + half * 16);
        fa.q[1] = *(const uint4*)(aptr + k + half * 16 + 8);
        fb.q[0] = *(const uint4*)(bptr + k);
        fb.q[1] = *(const uint4*)(bptr + k + 8);
        acc = __builtin_amdgcn_wmma_f32_16x16x32_f16(
            false, fa.v, false, fb.v, (short)0, acc, false, false);
    }

    const float scale = 0.08838834764831845f;  // 1/sqrt(128)
    int col = ktile * 16 + r;
    float madd = kpm[b * LP + col] ? -1e30f : 0.f;
    float* Sp = S + (size_t)bh * LP * LP;
#pragma unroll
    for (int v = 0; v < 8; ++v) {
        int row = qtile * 16 + half * 8 + v;
        Sp[(size_t)row * LP + col] = acc[v] * scale + madd;
    }
}

// one wave per row; writes P (f16, row stride LPK, zero-padded cols >= LP)
__global__ void k_attn_softmax(const float* __restrict__ S, _Float16* __restrict__ P) {
    int row = blockIdx.x % LP;
    int bh = blockIdx.x / LP;
    const float* Sp = S + ((size_t)bh * LP + row) * LP;
    _Float16* Pp = P + ((size_t)bh * LP + row) * LPK;
    int lane = threadIdx.x;

    float vals[5];
    float m = -3.4e38f;
#pragma unroll
    for (int i = 0; i < 5; ++i) {
        int c = lane + i * 32;
        vals[i] = (c < LP) ? Sp[c] : -3.4e38f;
        m = fmaxf(m, vals[i]);
    }
    for (int o = 16; o; o >>= 1) m = fmaxf(m, __shfl_xor(m, o, 32));
    float sum = 0.f;
#pragma unroll
    for (int i = 0; i < 5; ++i) {
        int c = lane + i * 32;
        float e = (c < LP) ? __expf(vals[i] - m) : 0.f;
        vals[i] = e;
        sum += e;
    }
    for (int o = 16; o; o >>= 1) sum += __shfl_xor(sum, o, 32);
    float inv = 1.f / sum;
#pragma unroll
    for (int i = 0; i < 5; ++i) {
        int c = lane + i * 32;
        if (c < LPK) Pp[c] = (_Float16)((c < LP) ? vals[i] * inv : 0.f);
    }
}

// O = P * V   (grid = B*4*9, block = 256 = 8 waves, wave = n-tile of dh=128)
__global__ void k_attn_pv(const _Float16* __restrict__ qkv,
                          const _Float16* __restrict__ P,
                          _Float16* __restrict__ O) {
    int lane = threadIdx.x & 31;
    int wave = threadIdx.x >> 5;      // 0..7
    int qtile = blockIdx.x % 9;
    int bh = blockIdx.x / 9;
    int h = bh & 3, b = bh >> 2;
    int half = lane >> 4, r = lane & 15;

    const _Float16* Pp = P + (size_t)bh * LP * LPK;
    const _Float16* aptr = Pp + (size_t)(qtile * 16 + r) * LPK;
    const _Float16* Vb = qkv + (size_t)(b * LP) * 1536 + 1024 + h * 128;
    int ncol = wave * 16 + r;

    v8f acc = {};
#pragma unroll
    for (int k = 0; k < LPK; k += 32) {
        FragH fa, fb;
        fa.q[0] = *(const uint4*)(aptr + k + half * 16);
        fa.q[1] = *(const uint4*)(aptr + k + half * 16 + 8);
#pragma unroll
        for (int i = 0; i < 16; ++i) {
            int kk = k + half * 16 + i;
            fb.e[i] = (kk < LP) ? Vb[(size_t)kk * 1536 + ncol] : (_Float16)0.f;
        }
        acc = __builtin_amdgcn_wmma_f32_16x16x32_f16(
            false, fa.v, false, fb.v, (short)0, acc, false, false);
    }
#pragma unroll
    for (int v = 0; v < 8; ++v) {
        int row = qtile * 16 + half * 8 + v;
        O[(size_t)(b * LP + row) * DD + h * 128 + wave * 16 + r] = (_Float16)acc[v];
    }
}

// ---------------- ring attention (VALU; 0.1% of FLOPs) ----------------
__global__ void k_ring_attn(const _Float16* __restrict__ qkv,
                            const int* __restrict__ rnn,
                            _Float16* __restrict__ O) {
    __shared__ float Sm[8][8];
    __shared__ float Pm[8][8];
    int r = blockIdx.x >> 1, h = blockIdx.x & 1;
    int n = rnn[r];
    const _Float16* base = qkv + (size_t)(r * 8) * 1536;

    int q = threadIdx.x >> 3, kk = threadIdx.x & 7;   // 64 threads
    {
        const _Float16* Qp = base + (size_t)q * 1536 + h * 256;
        const _Float16* Kp = base + (size_t)kk * 1536 + 512 + h * 256;
        float s = 0.f;
        for (int d = 0; d < 256; ++d) s += h2f(Qp[d]) * h2f(Kp[d]);
        s *= 0.0625f;                                  // 1/sqrt(256)
        if (kk >= n) s = -1e30f;
        Sm[q][kk] = s;
    }
    __syncthreads();
    if (threadIdx.x < 8) {
        int row = threadIdx.x;
        float m = Sm[row][0];
        for (int j = 1; j < 8; ++j) m = fmaxf(m, Sm[row][j]);
        float e[8], sum = 0.f;
        for (int j = 0; j < 8; ++j) { e[j] = __expf(Sm[row][j] - m); sum += e[j]; }
        float inv = 1.f / sum;
        for (int j = 0; j < 8; ++j) Pm[row][j] = e[j] * inv;
    }
    __syncthreads();
    for (int idx = threadIdx.x; idx < 8 * 256; idx += 64) {
        int j = idx >> 8, d = idx & 255;
        float o = 0.f;
        for (int k2 = 0; k2 < 8; ++k2)
            o += Pm[j][k2] * h2f(base[(size_t)k2 * 1536 + 1024 + h * 256 + d]);
        O[(size_t)(r * 8 + j) * DD + h * 256 + d] = (_Float16)o;
    }
}

// ---------------- residual + layernorm (in-place into res; also f16 copy) ----------
__global__ void k_ln_residual(float* __restrict__ res, const float* __restrict__ y,
                              const float* __restrict__ g, const float* __restrict__ bt,
                              _Float16* __restrict__ outh) {
    __shared__ float sred[8];
    size_t base = (size_t)blockIdx.x * DD;
    int i0 = threadIdx.x, i1 = threadIdx.x + 256;
    float a0 = res[base + i0] + y[base + i0];
    float a1 = res[base + i1] + y[base + i1];
    float s = a0 + a1;
    for (int o = 16; o; o >>= 1) s += __shfl_xor(s, o, 32);
    if ((threadIdx.x & 31) == 0) sred[threadIdx.x >> 5] = s;
    __syncthreads();
    float tot = 0.f;
    for (int w = 0; w < 8; ++w) tot += sred[w];
    float mu = tot * (1.f / DD);
    __syncthreads();
    float d0 = a0 - mu, d1 = a1 - mu;
    float vs = d0 * d0 + d1 * d1;
    for (int o = 16; o; o >>= 1) vs += __shfl_xor(vs, o, 32);
    if ((threadIdx.x & 31) == 0) sred[threadIdx.x >> 5] = vs;
    __syncthreads();
    float vtot = 0.f;
    for (int w = 0; w < 8; ++w) vtot += sred[w];
    float rstd = rsqrtf(vtot * (1.f / DD) + 1e-5f);
    float o0 = d0 * rstd * g[i0] + bt[i0];
    float o1 = d1 * rstd * g[i1] + bt[i1];
    res[base + i0] = o0;  res[base + i1] = o1;
    outh[base + i0] = (_Float16)o0;  outh[base + i1] = (_Float16)o1;
}

// ---------------- argmax-|.| pooling over 8 ring positions ----------------
__global__ void k_ring_pool(const float* __restrict__ h, float* __restrict__ rv) {
    int r = blockIdx.x, d = threadIdx.x;        // 512 threads
    size_t base = (size_t)r * 8 * DD + d;
    float best = h[base];
    float ba = fabsf(best);
    for (int j = 1; j < 8; ++j) {
        float v = h[base + (size_t)j * DD];
        float a = fabsf(v);
        if (a > ba) { ba = a; best = v; }
    }
    rv[(size_t)r * DD + d] = best;
}

// ---------------- output extraction ----------------
__global__ void k_out_mol(const float* __restrict__ seq, float* __restrict__ out) {
    int b = blockIdx.x, d = threadIdx.x;        // 512 threads
    out[(size_t)b * DD + d] = seq[((size_t)b * LP) * DD + d];
}

__global__ void k_out_node(const float* __restrict__ seq, const int* __restrict__ ptr,
                           float* __restrict__ out) {
    int b = blockIdx.x >> 7, i = blockIdx.x & 127;
    int nn = ptr[b + 1] - ptr[b];
    if (i < nn) {
        size_t dstrow = (size_t)BSZ + ptr[b] + i;
        out[dstrow * DD + threadIdx.x] = seq[((size_t)(b * LP) + 1 + i) * DD + threadIdx.x];
    }
}

__global__ void k_out_ring(const float* __restrict__ seq, const int* __restrict__ mrn,
                           const int* __restrict__ rstartB, float* __restrict__ out, int N) {
    int b = blockIdx.x >> 3, j = blockIdx.x & 7;
    if (j < mrn[b]) {
        size_t dstrow = (size_t)BSZ + (size_t)N + rstartB[b] + j;
        out[dstrow * DD + threadIdx.x] = seq[((size_t)(b * LP) + 130 + j) * DD + threadIdx.x];
    }
}

// =====================================================================
extern "C" void kernel_launch(void* const* d_in, const int* in_sizes, int n_in,
                              void* d_out, int out_size, void* d_ws, size_t ws_size,
                              hipStream_t stream) {
    // ---- inputs ----
    const float* x   = (const float*)d_in[0];
    const int*  ptr  = (const int*)d_in[1];
    const int*  rni  = (const int*)d_in[2];
    const int*  rnn  = (const int*)d_in[3];
    const int*  mrn  = (const int*)d_in[4];
    const float* CLS = (const float*)d_in[5];
    const float* RING = (const float*)d_in[6];
    const float* END = (const float*)d_in[7];

    const float* r_wqkv = (const float*)d_in[8];
    const float* r_bqkv = (const float*)d_in[9];
    const float* r_wo   = (const float*)d_in[10];
    const float* r_bo   = (const float*)d_in[11];
    const float* r_ln1g = (const float*)d_in[12];
    const float* r_ln1b = (const float*)d_in[13];
    const float* r_w1   = (const float*)d_in[14];
    const float* r_b1   = (const float*)d_in[15];
    const float* r_w2   = (const float*)d_in[16];
    const float* r_b2   = (const float*)d_in[17];
    const float* r_ln2g = (const float*)d_in[18];
    const float* r_ln2b = (const float*)d_in[19];
    const float* m_wqkv = (const float*)d_in[20];
    const float* m_bqkv = (const float*)d_in[21];
    const float* m_wo   = (const float*)d_in[22];
    const float* m_bo   = (const float*)d_in[23];
    const float* m_ln1g = (const float*)d_in[24];
    const float* m_ln1b = (const float*)d_in[25];
    const float* m_w1   = (const float*)d_in[26];
    const float* m_b1   = (const float*)d_in[27];
    const float* m_w2   = (const float*)d_in[28];
    const float* m_b2   = (const float*)d_in[29];
    const float* m_ln2g = (const float*)d_in[30];
    const float* m_ln2b = (const float*)d_in[31];

    const int N  = in_sizes[0] / DD;     // total nodes
    const int R  = in_sizes[3];          // total rings
    const int TA = R * LR;               // ring-stage tokens
    const int TB = BSZ * LP;             // molecule-stage tokens

    // ---- workspace layout ----
    char* wsp = (char*)d_ws;
    size_t off = 0;
    auto alloc = [&](size_t bytes) -> char* {
        char* p = wsp + off;
        off += (bytes + 255) & ~(size_t)255;
        return p;
    };

    // persistent
    _Float16* w_rqkv = (_Float16*)alloc((size_t)1536 * DD * 2);
    _Float16* w_rwo  = (_Float16*)alloc((size_t)DD * DD * 2);
    _Float16* w_rw1  = (_Float16*)alloc((size_t)FFD * DD * 2);
    _Float16* w_rw2  = (_Float16*)alloc((size_t)DD * FFD * 2);
    _Float16* w_mqkv = (_Float16*)alloc((size_t)1536 * DD * 2);
    _Float16* w_mwo  = (_Float16*)alloc((size_t)DD * DD * 2);
    _Float16* w_mw1  = (_Float16*)alloc((size_t)FFD * DD * 2);
    _Float16* w_mw2  = (_Float16*)alloc((size_t)DD * FFD * 2);
    int* rstartR = (int*)alloc((size_t)R * 4);
    int* rstartB = (int*)alloc((size_t)BSZ * 4);
    unsigned char* kpm = (unsigned char*)alloc((size_t)BSZ * LP);
    float* rings_vec = (float*)alloc((size_t)R * DD * 4);

    // arena (stage A and stage B reuse the same region)
    char* arena = wsp + off;
    size_t aoff = 0;
    auto aalloc = [&](size_t bytes) -> char* {
        char* p = arena + aoff;
        aoff += (bytes + 255) & ~(size_t)255;
        return p;
    };
    // stage A
    float*    a_h0  = (float*)aalloc((size_t)TA * DD * 4);
    _Float16* a_hb  = (_Float16*)aalloc((size_t)TA * DD * 2);
    _Float16* a_qkv = (_Float16*)aalloc((size_t)TA * 1536 * 2);
    _Float16* a_o   = (_Float16*)aalloc((size_t)TA * DD * 2);
    float*    a_p   = (float*)aalloc((size_t)TA * DD * 4);
    _Float16* a_ff  = (_Float16*)aalloc((size_t)TA * FFD * 2);
    // stage B (same base, used after rings_vec is produced)
    aoff = 0;
    float*    s_seq = (float*)aalloc((size_t)TB * DD * 4);
    _Float16* s_hb  = (_Float16*)aalloc((size_t)TB * DD * 2);
    _Float16* s_qkv = (_Float16*)aalloc((size_t)TB * 1536 * 2);
    float*    s_S   = (float*)aalloc((size_t)BSZ * 4 * LP * LP * 4);
    _Float16* s_P   = (_Float16*)aalloc((size_t)BSZ * 4 * LP * LPK * 2);
    _Float16* s_o   = (_Float16*)aalloc((size_t)TB * DD * 2);
    float*    s_p   = (float*)aalloc((size_t)TB * DD * 4);
    _Float16* s_ff  = (_Float16*)aalloc((size_t)TB * FFD * 2);

    float* out = (float*)d_out;

    auto cvt = [&](const float* src, _Float16* dst, size_t n) {
        k_f32_to_f16<<<(unsigned)((n + 255) / 256), 256, 0, stream>>>(src, dst, n);
    };

    // ---- weight conversion ----
    cvt(r_wqkv, w_rqkv, (size_t)1536 * DD);
    cvt(r_wo,   w_rwo,  (size_t)DD * DD);
    cvt(r_w1,   w_rw1,  (size_t)FFD * DD);
    cvt(r_w2,   w_rw2,  (size_t)DD * FFD);
    cvt(m_wqkv, w_mqkv, (size_t)1536 * DD);
    cvt(m_wo,   w_mwo,  (size_t)DD * DD);
    cvt(m_w1,   w_mw1,  (size_t)FFD * DD);
    cvt(m_w2,   w_mw2,  (size_t)DD * FFD);

    // ---- scans ----
    k_scan_excl<<<1, 1, 0, stream>>>(rnn, rstartR, R);
    k_scan_excl<<<1, 1, 0, stream>>>(mrn, rstartB, BSZ);

    // ================= Stage A: ring encoder =================
    k_build_ring_tokens<<<TA, DD, 0, stream>>>(x, rni, rnn, rstartR, a_h0, a_hb);

    dim3 blk(256);
    // qkv = hb @ Wqkv^T + b   (f16 out)
    k_gemm_wmma<false, true><<<dim3((TA + 127) / 128, 1536 / 64), blk, 0, stream>>>(
        a_hb, DD, w_rqkv, r_bqkv, nullptr, a_qkv, 1536, TA, DD);
    // attention (VALU, 8x8 scores)
    k_ring_attn<<<R * 2, 64, 0, stream>>>(a_qkv, rnn, a_o);
    // proj (f32 out)
    k_gemm_wmma<false, false><<<dim3((TA + 127) / 128, DD / 64), blk, 0, stream>>>(
        a_o, DD, w_rwo, r_bo, a_p, nullptr, DD, TA, DD);
    k_ln_residual<<<TA, 256, 0, stream>>>(a_h0, a_p, r_ln1g, r_ln1b, a_hb);
    // ffn
    k_gemm_wmma<true, true><<<dim3((TA + 127) / 128, FFD / 64), blk, 0, stream>>>(
        a_hb, DD, w_rw1, r_b1, nullptr, a_ff, FFD, TA, DD);
    k_gemm_wmma<false, false><<<dim3((TA + 127) / 128, DD / 64), blk, 0, stream>>>(
        a_ff, FFD, w_rw2, r_b2, a_p, nullptr, DD, TA, FFD);
    k_ln_residual<<<TA, 256, 0, stream>>>(a_h0, a_p, r_ln2g, r_ln2b, a_hb);
    // pool
    k_ring_pool<<<R, DD, 0, stream>>>(a_h0, rings_vec);

    // ================= Stage B: molecule encoder =================
    k_build_kpm<<<BSZ, 256, 0, stream>>>(ptr, mrn, kpm);
    k_build_seq<<<BSZ * LP, DD, 0, stream>>>(x, ptr, mrn, rstartB, CLS, RING, END,
                                             rings_vec, s_seq, s_hb);
    // qkv
    k_gemm_wmma<false, true><<<dim3(TB / 128, 1536 / 64), blk, 0, stream>>>(
        s_hb, DD, w_mqkv, m_bqkv, nullptr, s_qkv, 1536, TB, DD);
    // attention
    k_attn_scores<<<BSZ * 4 * 9, 288, 0, stream>>>(s_qkv, kpm, s_S);
    k_attn_softmax<<<BSZ * 4 * LP, 32, 0, stream>>>(s_S, s_P);
    k_attn_pv<<<BSZ * 4 * 9, 256, 0, stream>>>(s_qkv, s_P, s_o);
    // proj
    k_gemm_wmma<false, false><<<dim3(TB / 128, DD / 64), blk, 0, stream>>>(
        s_o, DD, w_mwo, m_bo, s_p, nullptr, DD, TB, DD);
    k_ln_residual<<<TB, 256, 0, stream>>>(s_seq, s_p, m_ln1g, m_ln1b, s_hb);
    // ffn
    k_gemm_wmma<true, true><<<dim3(TB / 128, FFD / 64), blk, 0, stream>>>(
        s_hb, DD, w_mw1, m_b1, nullptr, s_ff, FFD, TB, DD);
    k_gemm_wmma<false, false><<<dim3(TB / 128, DD / 64), blk, 0, stream>>>(
        s_ff, FFD, w_mw2, m_b2, s_p, nullptr, DD, TB, FFD);
    k_ln_residual<<<TB, 256, 0, stream>>>(s_seq, s_p, m_ln2g, m_ln2b, s_hb);

    // ---- outputs ----
    k_out_mol<<<BSZ, DD, 0, stream>>>(s_seq, out);
    k_out_node<<<BSZ * 128, DD, 0, stream>>>(s_seq, ptr, out);
    k_out_ring<<<BSZ * 8, DD, 0, stream>>>(s_seq, mrn, rstartB, out, N);
}
